// SemanticRoutedTransformerLayer_69784628625427
// MI455X (gfx1250) — compile-verified, matrix operand order checked by reading, not run on hardware
//
#include <hip/hip_runtime.h>
#include <hip/hip_bf16.h>

typedef __attribute__((ext_vector_type(16))) __bf16 v16bf;
typedef __attribute__((ext_vector_type(8)))  __bf16 v8bf;
typedef __attribute__((ext_vector_type(8)))  float  v8f;

#define DEVINL static __device__ __forceinline__

// ---- problem constants (match reference) ----
#define BB    4
#define SS    1024
#define HH    1024
#define II    4096
#define PPATH 16
#define TOPK  4
#define SDIM  256
#define NCAT  16
#define NHEAD 16
#define NHSEM 4
#define NTOK  (BB * SS)       /* 4096 tokens */
#define RINP  288             /* 272 padded to multiple of 32 */

// ======================= WMMA helpers =======================

DEVINL v8f wmma_bf16(v16bf a, v16bf b, v8f c) {
  return __builtin_amdgcn_wmma_f32_16x16x32_bf16(false, a, false, b, (short)0, c,
                                                 false, false);
}

DEVINL int frag_kk(int v, int half) {
  return (v < 4 ? 2 * v : 16 + 2 * (v - 4)) + 8 * half;
}

// Row-major fragment load: element pairs for one lane form two contiguous
// 8-element (16B) runs at k0+8*half and k0+16+8*half.  Used for A fragments
// and for B fragments accessed transposed (B[n][k]).  No bounds checks:
// caller guarantees K % 32 == 0 and 16B alignment.
DEVINL v16bf load_frag(const __bf16* __restrict__ P, int ld, int base, int k0) {
  int lane = threadIdx.x & 31;
  int r = lane & 15, h = lane >> 4;
  const __bf16* p = P + (size_t)(base + r) * (size_t)ld + k0 + 8 * h;
  v8bf lo = *(const v8bf*)p;
  v8bf hi = *(const v8bf*)(p + 16);
  v16bf f;
#pragma unroll
  for (int i = 0; i < 8; ++i) { f[i] = lo[i]; f[i + 8] = hi[i]; }
  return f;
}

template <int ACT>
DEVINL float act_fn(float x) {
  if (ACT == 1) return fmaxf(x, 0.0f);                       // relu
  if (ACT == 2) {                                            // tanh-gelu
    float x3 = x * x * x;
    return 0.5f * x * (1.0f + tanhf(0.7978845608028654f * (x + 0.044715f * x3)));
  }
  if (ACT == 3) return 1.0f / (1.0f + __expf(-x));           // sigmoid
  return x;
}

// ======================= generic WMMA GEMM (register-blocked) =======================
// C[z] = act(alpha * A[z] * Bt[z]^T + bias[z]) + beta * Cprev
// A: [M][lda] bf16 row-major; Bt: [N][ldb] bf16 (transposed weights).
// Each wave computes a (16*MT) x (16*NT) tile: MT+NT fragment loads feed
// MT*NT WMMAs per k-step (2x2 => 2 b128 loads per WMMA).
// All epilogue behavior is compile-time: HAS_BIAS / HAS_BETA / OUTF / OUTB.
// CBT=1: store bf16 result transposed as Cb[n][m] (ldc = column stride).
template <int ACT, bool CBT, int MT, int NT, bool HAS_BIAS, bool HAS_BETA,
          bool OUTF, bool OUTB>
__global__ void __launch_bounds__(256)
gemm_wmma_bf16(const __bf16* __restrict__ A, int lda, long sAz,
               const __bf16* __restrict__ Bt, int ldb, long sBz,
               float* __restrict__ Cf, __bf16* __restrict__ Cb, int ldc, long sCz,
               const float* __restrict__ bias, long sBiasz,
               int M, int N, int Kdim, float alpha, float beta) {
  int wave = threadIdx.x >> 5;
  int m0 = blockIdx.y * (128 * MT) + wave * (16 * MT);
  int n0 = blockIdx.x * (16 * NT);
  if (m0 >= M || n0 >= N) return;
  A  += (size_t)blockIdx.z * sAz;
  Bt += (size_t)blockIdx.z * sBz;
  size_t czoff = (size_t)blockIdx.z * sCz;

  v8f acc[MT][NT] = {};
  for (int k0 = 0; k0 < Kdim; k0 += 32) {
    v16bf af[MT], bf[NT];
#pragma unroll
    for (int mt = 0; mt < MT; ++mt) af[mt] = load_frag(A, lda, m0 + 16 * mt, k0);
#pragma unroll
    for (int nt = 0; nt < NT; ++nt) bf[nt] = load_frag(Bt, ldb, n0 + 16 * nt, k0);
#pragma unroll
    for (int mt = 0; mt < MT; ++mt)
#pragma unroll
      for (int nt = 0; nt < NT; ++nt)
        acc[mt][nt] = wmma_bf16(af[mt], bf[nt], acc[mt][nt]);
  }

  int lane = threadIdx.x & 31;
  int r = lane & 15, h = lane >> 4;

#pragma unroll
  for (int mt = 0; mt < MT; ++mt) {
#pragma unroll
    for (int nt = 0; nt < NT; ++nt) {
      int col = n0 + 16 * nt + r;
      int rb = m0 + 16 * mt;
      float bv = 0.0f;
      if (HAS_BIAS) bv = bias[(size_t)blockIdx.z * sBiasz + col];
      float vals[8];
#pragma unroll
      for (int i = 0; i < 8; ++i) {
        float v = act_fn<ACT>(alpha * acc[mt][nt][i] + bv);
        if (HAS_BETA)
          v += beta * Cf[czoff + (size_t)(rb + i + 8 * h) * ldc + col];
        vals[i] = v;
      }
      if (CBT) {
        v8bf pack;
#pragma unroll
        for (int i = 0; i < 8; ++i) pack[i] = (__bf16)vals[i];
        *(v8bf*)(Cb + czoff + (size_t)col * ldc + rb + 8 * h) = pack;
      } else {
#pragma unroll
        for (int i = 0; i < 8; ++i) {
          size_t idx = czoff + (size_t)(rb + i + 8 * h) * ldc + col;
          if (OUTF) Cf[idx] = vals[i];
          if (OUTB) Cb[idx] = (__bf16)vals[i];
        }
      }
    }
  }
}

// ======================= fused flash attention (head dim 64) =======================
// Q,K token-major bf16; V transposed [dim][token] bf16; O token-major bf16.
// grid = (S/16, nheads, B), block = 32 (one wave per 16-query tile).
// Key tiles processed 32 at a time (dense K=32 WMMAs throughout).
template <bool CAUSAL>
__global__ void __launch_bounds__(32)
flash_attn_wmma(const __bf16* __restrict__ Qb, int rsQ,
                const __bf16* __restrict__ Kb, int rsK,
                const __bf16* __restrict__ VT, int ldv,
                __bf16* __restrict__ O, int rsO, int S) {
  int qt = blockIdx.x, hd = blockIdx.y, b = blockIdx.z;
  int lane = threadIdx.x & 31;
  int r = lane & 15, hf = lane >> 4;

  const __bf16* Q  = Qb + (size_t)b * S * rsQ + hd * 64;
  const __bf16* Kp = Kb + (size_t)b * S * rsK + hd * 64;
  const __bf16* Vt = VT + (size_t)(hd * 64) * ldv + (size_t)b * S;

  v16bf qa0 = load_frag(Q, rsQ, qt * 16, 0);
  v16bf qa1 = load_frag(Q, rsQ, qt * 16, 32);

  v8f cc0 = {}, cc1 = {}, cc2 = {}, cc3 = {};
  float m[8], l[8];
#pragma unroll
  for (int i = 0; i < 8; ++i) { m[i] = -3.0e38f; l[i] = 0.0f; }

  __shared__ float pt[16][33];

  int ktend = CAUSAL ? (qt >> 1) : (S / 32 - 1);
  for (int kt = 0; kt <= ktend; ++kt) {
    int c0 = kt * 32;
    // two 16-col score tiles = Q * K^T over the 32-key window
    v8f s0 = {}, s1 = {};
    s0 = wmma_bf16(qa0, load_frag(Kp, rsK, c0, 0), s0);
    s0 = wmma_bf16(qa1, load_frag(Kp, rsK, c0, 32), s0);
    s1 = wmma_bf16(qa0, load_frag(Kp, rsK, c0 + 16, 0), s1);
    s1 = wmma_bf16(qa1, load_frag(Kp, rsK, c0 + 16, 32), s1);

    float pr0[8], pr1[8];
#pragma unroll
    for (int i = 0; i < 8; ++i) {
      int row = qt * 16 + i + 8 * hf;
      float x0 = s0[i] * 0.125f;  // 1/sqrt(64)
      float x1 = s1[i] * 0.125f;
      if (CAUSAL) {
        if (c0 + r > row)      x0 = -3.0e38f;
        if (c0 + 16 + r > row) x1 = -3.0e38f;
      }
      float mx = fmaxf(x0, x1);
      mx = fmaxf(mx, __shfl_xor(mx, 1, 32));
      mx = fmaxf(mx, __shfl_xor(mx, 2, 32));
      mx = fmaxf(mx, __shfl_xor(mx, 4, 32));
      mx = fmaxf(mx, __shfl_xor(mx, 8, 32));
      float mn = fmaxf(m[i], mx);
      float p0 = __expf(x0 - mn);
      float p1 = __expf(x1 - mn);
      float rs = p0 + p1;
      rs += __shfl_xor(rs, 1, 32);
      rs += __shfl_xor(rs, 2, 32);
      rs += __shfl_xor(rs, 4, 32);
      rs += __shfl_xor(rs, 8, 32);
      float al = __expf(m[i] - mn);
      l[i] = l[i] * al + rs;
      m[i] = mn;
      cc0[i] *= al; cc1[i] *= al; cc2[i] *= al; cc3[i] *= al;
      pr0[i] = p0; pr1[i] = p1;
    }

    // re-layout P (C layout, 16x32) -> A fragment via LDS
    __syncthreads();
#pragma unroll
    for (int i = 0; i < 8; ++i) {
      pt[i + 8 * hf][r]      = pr0[i];
      pt[i + 8 * hf][16 + r] = pr1[i];
    }
    __syncthreads();
    v16bf pa;
#pragma unroll
    for (int v = 0; v < 8; ++v) {
      int kk = frag_kk(v, hf);
      pa[2 * v]     = (__bf16)pt[r][kk];
      pa[2 * v + 1] = (__bf16)pt[r][kk + 1];
    }

    // O += P * V ; V^T fragments are contiguous along the token (K) axis
    cc0 = wmma_bf16(pa, load_frag(Vt, ldv, 0, c0),  cc0);
    cc1 = wmma_bf16(pa, load_frag(Vt, ldv, 16, c0), cc1);
    cc2 = wmma_bf16(pa, load_frag(Vt, ldv, 32, c0), cc2);
    cc3 = wmma_bf16(pa, load_frag(Vt, ldv, 48, c0), cc3);
  }

  __bf16* Op = O + (size_t)b * S * rsO + hd * 64;
#pragma unroll
  for (int i = 0; i < 8; ++i) {
    float inv = l[i] > 0.0f ? 1.0f / l[i] : 0.0f;
    size_t base = (size_t)(qt * 16 + i + 8 * hf) * rsO;
    Op[base + 0 + r]  = (__bf16)(cc0[i] * inv);
    Op[base + 16 + r] = (__bf16)(cc1[i] * inv);
    Op[base + 32 + r] = (__bf16)(cc2[i] * inv);
    Op[base + 48 + r] = (__bf16)(cc3[i] * inv);
  }
}

// ======================= LayerNorm (row-wise, block=256) =======================
template <bool RELU>
__global__ void __launch_bounds__(256)
layernorm_k(const float* __restrict__ x, const float* __restrict__ g,
            const float* __restrict__ b, __bf16* __restrict__ ob, int D) {
  int row = blockIdx.x;
  const float* xr = x + (size_t)row * D;
  __shared__ float red[256];
  int tid = threadIdx.x;

  float s = 0.0f;
  for (int j = tid; j < D; j += 256) s += xr[j];
  red[tid] = s; __syncthreads();
  for (int st = 128; st > 0; st >>= 1) {
    if (tid < st) red[tid] += red[tid + st];
    __syncthreads();
  }
  float mean = red[0] / (float)D;
  __syncthreads();

  float vs = 0.0f;
  for (int j = tid; j < D; j += 256) { float d = xr[j] - mean; vs += d * d; }
  red[tid] = vs; __syncthreads();
  for (int st = 128; st > 0; st >>= 1) {
    if (tid < st) red[tid] += red[tid + st];
    __syncthreads();
  }
  float rstd = rsqrtf(red[0] / (float)D + 1e-5f);

  for (int j = tid; j < D; j += 256) {
    float v = (xr[j] - mean) * rstd * g[j] + b[j];
    if (RELU) v = fmaxf(v, 0.0f);
    ob[(size_t)row * D + j] = (__bf16)v;
  }
}

// ======================= convert / transpose kernels =======================
__global__ void f32_to_bf16_k(const float* __restrict__ s, __bf16* __restrict__ d, size_t n) {
  size_t i = (size_t)blockIdx.x * blockDim.x + threadIdx.x;
  if (i < n) d[i] = (__bf16)s[i];
}

// dst[n][k] = src[k][n] (bf16), with zero padding for k in [K, ldd)
__global__ void transpose_cvt_k(const float* __restrict__ src, int lds, long sSz,
                                __bf16* __restrict__ dst, int ldd, long sDz,
                                int K, int N) {
  int k = blockIdx.x * 32 + (threadIdx.x & 31);
  int n = blockIdx.y * 8 + (threadIdx.x >> 5);
  if (k >= ldd || n >= N) return;
  float v = (k < K) ? src[(size_t)blockIdx.z * sSz + (size_t)k * lds + n] : 0.0f;
  dst[(size_t)blockIdx.z * sDz + (size_t)n * ldd + k] = (__bf16)v;
}

// ======================= small utility kernels =======================
__global__ void zero_k(float* p, int n) {
  int i = blockIdx.x * blockDim.x + threadIdx.x;
  if (i < n) p[i] = 0.0f;
}

__global__ void concat_rin_k(const __bf16* __restrict__ ctx, const float* __restrict__ cat,
                             __bf16* __restrict__ rin) {
  int row = blockIdx.x, j = threadIdx.x;
  if (j < SDIM)             rin[(size_t)row * RINP + j] = ctx[(size_t)row * SDIM + j];
  else if (j < SDIM + NCAT) rin[(size_t)row * RINP + j] = (__bf16)cat[(size_t)row * NCAT + (j - SDIM)];
  else if (j < RINP)        rin[(size_t)row * RINP + j] = (__bf16)0.0f;
}

__global__ void topk_route_k(const float* __restrict__ scores, float* __restrict__ w, int rows) {
  int row = blockIdx.x * blockDim.x + threadIdx.x;
  if (row >= rows) return;
  float s[PPATH];
  float mx = -3.0e38f;
  for (int j = 0; j < PPATH; ++j) { s[j] = scores[(size_t)row * PPATH + j]; mx = fmaxf(mx, s[j]); }
  float sum = 0.0f;
  for (int j = 0; j < PPATH; ++j) { s[j] = __expf(s[j] - mx); sum += s[j]; }
  float inv = 1.0f / sum;
  for (int j = 0; j < PPATH; ++j) s[j] *= inv;
  float sel[PPATH]; for (int j = 0; j < PPATH; ++j) sel[j] = 0.0f;
  float wsum = 0.0f;
  for (int t = 0; t < TOPK; ++t) {
    int bi = 0; float bv = -1.0f;
    for (int j = 0; j < PPATH; ++j)
      if (sel[j] == 0.0f && s[j] > bv) { bv = s[j]; bi = j; }
    sel[bi] = s[bi]; wsum += s[bi];
  }
  float winv = 1.0f / (wsum + 1e-8f);
  for (int j = 0; j < PPATH; ++j) w[(size_t)row * PPATH + j] = sel[j] * winv;
}

__global__ void glbl_accum_k(const float* __restrict__ scores, float* __restrict__ freq, int rows) {
  int row = blockIdx.x * blockDim.x + threadIdx.x;
  if (row >= rows) return;
  float s[PPATH];
  float mx = -3.0e38f;
  for (int j = 0; j < PPATH; ++j) { s[j] = scores[(size_t)row * PPATH + j]; mx = fmaxf(mx, s[j]); }
  float sum = 0.0f;
  for (int j = 0; j < PPATH; ++j) { s[j] = __expf(s[j] - mx); sum += s[j]; }
  float sc = 1.0f / (sum * (float)rows);
  for (int j = 0; j < PPATH; ++j) atomicAdd(&freq[j], s[j] * sc);
}

__global__ void glbl_final_k(const float* __restrict__ freq, float* __restrict__ out) {
  if (threadIdx.x == 0 && blockIdx.x == 0) {
    float acc = 0.0f;
    for (int j = 0; j < PPATH; ++j) acc += freq[j] * freq[j];
    out[0] = (float)PPATH * acc;
  }
}

__global__ void combine_out_k(const float* __restrict__ x, const float* __restrict__ ob,
                              const float* __restrict__ w, float* __restrict__ out) {
  size_t i = (size_t)blockIdx.x * blockDim.x + threadIdx.x;
  if (i >= (size_t)NTOK * HH) return;
  size_t row = i / HH;
  int h = (int)(i % HH);
  out[i] = x[i] + ob[i] * w[row * PPATH + (h >> 6)];
}

// ======================= host driver =======================
extern "C" void kernel_launch(void* const* d_in, const int* in_sizes, int n_in,
                              void* d_out, int out_size, void* d_ws, size_t ws_size,
                              hipStream_t stream) {
  (void)in_sizes; (void)n_in; (void)out_size; (void)ws_size;

  const float* hidden  = (const float*)d_in[0];
  const float* ln1_g   = (const float*)d_in[1];
  const float* ln1_b   = (const float*)d_in[2];
  const float* Wqkv    = (const float*)d_in[3];
  const float* bqkv    = (const float*)d_in[4];
  const float* Wao     = (const float*)d_in[5];
  const float* bao     = (const float*)d_in[6];
  const float* ln2_g   = (const float*)d_in[7];
  const float* ln2_b   = (const float*)d_in[8];
  const float* Wfc     = (const float*)d_in[9];
  const float* bfc     = (const float*)d_in[10];
  const float* Wpr     = (const float*)d_in[11];
  const float* bpr     = (const float*)d_in[12];
  const float* sem_W1  = (const float*)d_in[13];
  const float* sem_b1  = (const float*)d_in[14];
  const float* sem_lng = (const float*)d_in[15];
  const float* sem_lnb = (const float*)d_in[16];
  const float* sem_W2  = (const float*)d_in[17];
  const float* sem_b2  = (const float*)d_in[18];
  const float* mha_Wq  = (const float*)d_in[19];
  const float* mha_Wk  = (const float*)d_in[20];
  const float* mha_Wv  = (const float*)d_in[21];
  const float* mha_bq  = (const float*)d_in[22];
  const float* mha_bk  = (const float*)d_in[23];
  const float* mha_bv  = (const float*)d_in[24];
  const float* mha_Wo  = (const float*)d_in[25];
  const float* mha_bo  = (const float*)d_in[26];
  const float* cat_W   = (const float*)d_in[27];
  const float* cat_b   = (const float*)d_in[28];
  const float* rt_W1   = (const float*)d_in[29];
  const float* rt_b1   = (const float*)d_in[30];
  const float* rt_W2   = (const float*)d_in[31];
  const float* rt_b2   = (const float*)d_in[32];
  const float* path_emb = (const float*)d_in[33];

  float* out = (float*)d_out;

  // ---- workspace bump allocator ----
  char* base = (char*)d_ws;
  size_t off = 0;
  auto alloc = [&](size_t bytes) -> void* {
    void* p = base + off;
    off += (bytes + 255) & ~(size_t)255;
    return p;
  };

  // bf16 transposed weights  [N][Kpad]
  __bf16* wqkv_t = (__bf16*)alloc((size_t)3 * HH * HH * 2);   // [3072][1024]
  __bf16* wao_t  = (__bf16*)alloc((size_t)HH * HH * 2);       // [1024][1024]
  __bf16* wfc_t  = (__bf16*)alloc((size_t)PPATH * 256 * 64 * 2);  // [16][256][64]
  __bf16* wpr_t  = (__bf16*)alloc((size_t)PPATH * 64 * 256 * 2);  // [16][64][256]
  __bf16* sw1_t  = (__bf16*)alloc((size_t)SDIM * HH * 2);     // [256][1024]
  __bf16* sw2_t  = (__bf16*)alloc((size_t)SDIM * SDIM * 2);
  __bf16* wq_t   = (__bf16*)alloc((size_t)SDIM * SDIM * 2);
  __bf16* wk_t   = (__bf16*)alloc((size_t)SDIM * SDIM * 2);
  __bf16* wv_t   = (__bf16*)alloc((size_t)SDIM * SDIM * 2);
  __bf16* wo_t   = (__bf16*)alloc((size_t)SDIM * SDIM * 2);
  __bf16* catw_t = (__bf16*)alloc((size_t)NCAT * SDIM * 2);   // already [N][K]
  __bf16* rtw1_t = (__bf16*)alloc((size_t)SDIM * RINP * 2);   // [256][288] padded
  __bf16* rtw2_t = (__bf16*)alloc((size_t)PPATH * SDIM * 2);  // [16][256]
  __bf16* pe_t   = (__bf16*)alloc((size_t)PPATH * SDIM * 2);  // already [N][K]

  // activations
  __bf16* ln1b   = (__bf16*)alloc((size_t)NTOK * HH * 2);
  __bf16* qkb    = (__bf16*)alloc((size_t)NTOK * 2 * HH * 2); // q|k token-major
  __bf16* vtb    = (__bf16*)alloc((size_t)HH * NTOK * 2);     // v^T [1024][NTOK]
  __bf16* attnb  = (__bf16*)alloc((size_t)NTOK * HH * 2);
  float*  xf     = (float*) alloc((size_t)NTOK * HH * 4);
  __bf16* h2b    = (__bf16*)alloc((size_t)NTOK * HH * 2);
  float*  t1f    = (float*) alloc((size_t)NTOK * SDIM * 4);
  __bf16* t1rb   = (__bf16*)alloc((size_t)NTOK * SDIM * 2);
  __bf16* sfb    = (__bf16*)alloc((size_t)NTOK * SDIM * 2);
  __bf16* qsb    = (__bf16*)alloc((size_t)NTOK * SDIM * 2);
  __bf16* ksb    = (__bf16*)alloc((size_t)NTOK * SDIM * 2);
  __bf16* vstb   = (__bf16*)alloc((size_t)SDIM * NTOK * 2);   // v^T [256][NTOK]
  __bf16* sctxb  = (__bf16*)alloc((size_t)NTOK * SDIM * 2);
  __bf16* ctxb   = (__bf16*)alloc((size_t)NTOK * SDIM * 2);
  float*  catf   = (float*) alloc((size_t)NTOK * NCAT * 4);
  __bf16* rinb   = (__bf16*)alloc((size_t)NTOK * RINP * 2);
  __bf16* r1b    = (__bf16*)alloc((size_t)NTOK * SDIM * 2);
  float*  scoresf= (float*) alloc((size_t)NTOK * PPATH * 4);
  float*  wf     = (float*) alloc((size_t)NTOK * PPATH * 4);
  float*  freqf  = (float*) alloc(64);
  __bf16* interb = (__bf16*)alloc((size_t)NTOK * II * 2);
  float*  obf    = (float*) alloc((size_t)NTOK * HH * 4);

  // ---- weight conversion (transpose to [N][Kpad] bf16) ----
  auto tcvt = [&](const float* s, int lds, long sSz, __bf16* d, int ldd, long sDz,
                  int K, int N, int Z) {
    transpose_cvt_k<<<dim3(ldd / 32, N / 8, Z), 256, 0, stream>>>(s, lds, sSz, d, ldd, sDz, K, N);
  };
  tcvt(Wqkv, 3 * HH, 0, wqkv_t, HH, 0, HH, 3 * HH, 1);
  tcvt(Wao, HH, 0, wao_t, HH, 0, HH, HH, 1);
  tcvt(Wfc, II, (long)64 * II + 256, wfc_t, 64, 256 * 64, 64, 256, PPATH);
  tcvt(Wpr, HH, (long)256 * HH + 64, wpr_t, 256, 64 * 256, 256, 64, PPATH);
  tcvt(sem_W1, SDIM, 0, sw1_t, HH, 0, HH, SDIM, 1);
  tcvt(sem_W2, SDIM, 0, sw2_t, SDIM, 0, SDIM, SDIM, 1);
  tcvt(mha_Wq, SDIM, 0, wq_t, SDIM, 0, SDIM, SDIM, 1);
  tcvt(mha_Wk, SDIM, 0, wk_t, SDIM, 0, SDIM, SDIM, 1);
  tcvt(mha_Wv, SDIM, 0, wv_t, SDIM, 0, SDIM, SDIM, 1);
  tcvt(mha_Wo, SDIM, 0, wo_t, SDIM, 0, SDIM, SDIM, 1);
  tcvt(rt_W1, SDIM, 0, rtw1_t, RINP, 0, SDIM + NCAT, SDIM, 1);
  tcvt(rt_W2, PPATH, 0, rtw2_t, SDIM, 0, SDIM, PPATH, 1);
  // cat_W [16][256] and path_emb [16][256] are already [N][K] row-major
  f32_to_bf16_k<<<dim3((NCAT * SDIM + 255) / 256), 256, 0, stream>>>(cat_W, catw_t, (size_t)NCAT * SDIM);
  f32_to_bf16_k<<<dim3((PPATH * SDIM + 255) / 256), 256, 0, stream>>>(path_emb, pe_t, (size_t)PPATH * SDIM);

  const int GY2 = NTOK / 256;  // 16 (MT=2: 256 rows per block)

  // ---- base attention sublayer ----
  layernorm_k<false><<<NTOK, 256, 0, stream>>>(hidden, ln1_g, ln1_b, ln1b, HH);

  // q|k projection (token-major)  [bias, ->bf16]
  gemm_wmma_bf16<0, false, 2, 2, true, false, false, true>
      <<<dim3(2 * HH / 32, GY2, 1), 256, 0, stream>>>(
      ln1b, HH, 0, wqkv_t, HH, 0, nullptr, qkb, 2 * HH, 0, bqkv, 0,
      NTOK, 2 * HH, HH, 1.0f, 0.0f);
  // v projection, stored transposed [dim][token]
  gemm_wmma_bf16<0, true, 2, 2, true, false, false, true>
      <<<dim3(HH / 32, GY2, 1), 256, 0, stream>>>(
      ln1b, HH, 0, wqkv_t + (size_t)2 * HH * HH, HH, 0, nullptr, vtb, NTOK, 0,
      bqkv + 2 * HH, 0, NTOK, HH, HH, 1.0f, 0.0f);

  flash_attn_wmma<true><<<dim3(SS / 16, NHEAD, BB), 32, 0, stream>>>(
      qkb + 0, 2 * HH, qkb + HH, 2 * HH, vtb, NTOK, attnb, HH, SS);

  hipMemcpyAsync(xf, hidden, (size_t)NTOK * HH * 4, hipMemcpyDeviceToDevice, stream);

  // x = residual + attn@Wao + bao   [bias, beta, ->f32]
  gemm_wmma_bf16<0, false, 2, 2, true, true, true, false>
      <<<dim3(HH / 32, GY2, 1), 256, 0, stream>>>(
      attnb, HH, 0, wao_t, HH, 0, xf, nullptr, HH, 0, bao, 0,
      NTOK, HH, HH, 1.0f, 1.0f);

  // ---- semantic routing stack ----
  layernorm_k<false><<<NTOK, 256, 0, stream>>>(xf, ln2_g, ln2_b, h2b, HH);

  // t1 = h2 @ sem_W1 + b1   [bias, ->f32]
  gemm_wmma_bf16<0, false, 2, 2, true, false, true, false>
      <<<dim3(SDIM / 32, GY2, 1), 256, 0, stream>>>(
      h2b, HH, 0, sw1_t, HH, 0, t1f, nullptr, SDIM, 0, sem_b1, 0,
      NTOK, SDIM, HH, 1.0f, 0.0f);

  layernorm_k<true><<<NTOK, 256, 0, stream>>>(t1f, sem_lng, sem_lnb, t1rb, SDIM);

  // sf = t1r @ sem_W2 + b2   [bias, ->bf16]
  gemm_wmma_bf16<0, false, 2, 2, true, false, false, true>
      <<<dim3(SDIM / 32, GY2, 1), 256, 0, stream>>>(
      t1rb, SDIM, 0, sw2_t, SDIM, 0, nullptr, sfb, SDIM, 0, sem_b2, 0,
      NTOK, SDIM, SDIM, 1.0f, 0.0f);

  gemm_wmma_bf16<0, false, 2, 2, true, false, false, true>
      <<<dim3(SDIM / 32, GY2, 1), 256, 0, stream>>>(
      sfb, SDIM, 0, wq_t, SDIM, 0, nullptr, qsb, SDIM, 0, mha_bq, 0,
      NTOK, SDIM, SDIM, 1.0f, 0.0f);
  gemm_wmma_bf16<0, false, 2, 2, true, false, false, true>
      <<<dim3(SDIM / 32, GY2, 1), 256, 0, stream>>>(
      sfb, SDIM, 0, wk_t, SDIM, 0, nullptr, ksb, SDIM, 0, mha_bk, 0,
      NTOK, SDIM, SDIM, 1.0f, 0.0f);
  gemm_wmma_bf16<0, true, 2, 2, true, false, false, true>
      <<<dim3(SDIM / 32, GY2, 1), 256, 0, stream>>>(
      sfb, SDIM, 0, wv_t, SDIM, 0, nullptr, vstb, NTOK, 0, mha_bv, 0,
      NTOK, SDIM, SDIM, 1.0f, 0.0f);

  flash_attn_wmma<false><<<dim3(SS / 16, NHSEM, BB), 32, 0, stream>>>(
      qsb, SDIM, ksb, SDIM, vstb, NTOK, sctxb, SDIM, SS);

  gemm_wmma_bf16<0, false, 2, 2, true, false, false, true>
      <<<dim3(SDIM / 32, GY2, 1), 256, 0, stream>>>(
      sctxb, SDIM, 0, wo_t, SDIM, 0, nullptr, ctxb, SDIM, 0, mha_bo, 0,
      NTOK, SDIM, SDIM, 1.0f, 0.0f);

  // cat = sigmoid(ctx @ cat_W^T + cat_b)   [bias, ->f32]
  gemm_wmma_bf16<3, false, 2, 1, true, false, true, false>
      <<<dim3(NCAT / 16, GY2, 1), 256, 0, stream>>>(
      ctxb, SDIM, 0, catw_t, SDIM, 0, catf, nullptr, NCAT, 0, cat_b, 0,
      NTOK, NCAT, SDIM, 1.0f, 0.0f);

  concat_rin_k<<<NTOK, RINP, 0, stream>>>(ctxb, catf, rinb);

  // r1 = relu(rin @ rt_W1 + rt_b1)   [bias, ->bf16]
  gemm_wmma_bf16<1, false, 2, 2, true, false, false, true>
      <<<dim3(SDIM / 32, GY2, 1), 256, 0, stream>>>(
      rinb, RINP, 0, rtw1_t, RINP, 0, nullptr, r1b, SDIM, 0, rt_b1, 0,
      NTOK, SDIM, RINP, 1.0f, 0.0f);

  // scores = r1 @ rt_W2 + rt_b2   [bias, ->f32]
  gemm_wmma_bf16<0, false, 2, 1, true, false, true, false>
      <<<dim3(PPATH / 16, GY2, 1), 256, 0, stream>>>(
      r1b, SDIM, 0, rtw2_t, SDIM, 0, scoresf, nullptr, PPATH, 0, rt_b2, 0,
      NTOK, PPATH, SDIM, 1.0f, 0.0f);

  // scores += 0.1 * ctx @ path_emb^T   [beta, ->f32]
  gemm_wmma_bf16<0, false, 2, 1, false, true, true, false>
      <<<dim3(PPATH / 16, GY2, 1), 256, 0, stream>>>(
      ctxb, SDIM, 0, pe_t, SDIM, 0, scoresf, nullptr, PPATH, 0, nullptr, 0,
      NTOK, PPATH, SDIM, 0.1f, 1.0f);

  // GLBL loss
  zero_k<<<1, 32, 0, stream>>>(freqf, PPATH);
  glbl_accum_k<<<NTOK / 256, 256, 0, stream>>>(scoresf, freqf, NTOK);
  glbl_final_k<<<1, 32, 0, stream>>>(freqf, out + (size_t)NTOK * HH);

  // top-k routing weights
  topk_route_k<<<NTOK / 256, 256, 0, stream>>>(scoresf, wf, NTOK);

  // pathway-decomposed MLP (z = pathway index)
  // inter = gelu(h2_blk @ Wfc_blk + bfc_blk)   [bias, ->bf16]
  gemm_wmma_bf16<2, false, 2, 2, true, false, false, true>
      <<<dim3((II / PPATH) / 32, GY2, PPATH), 256, 0, stream>>>(
      h2b, HH, 64, wfc_t, 64, (long)256 * 64, nullptr, interb, II, 256,
      bfc, 256, NTOK, II / PPATH, 64, 1.0f, 0.0f);

  // ob = inter_blk @ Wpr_blk + bpr_blk   [bias, ->f32]
  gemm_wmma_bf16<0, false, 2, 2, true, false, true, false>
      <<<dim3((HH / PPATH) / 32, GY2, PPATH), 256, 0, stream>>>(
      interb, II, 256, wpr_t, 256, (long)64 * 256, obf, nullptr, HH, 64,
      bpr, 64, NTOK, HH / PPATH, 256, 1.0f, 0.0f);

  // out = x + w * ob
  combine_out_k<<<dim3((unsigned)(((size_t)NTOK * HH) / 256)), 256, 0, stream>>>(
      xf, obf, wf, out);
}
